// Transformer_1589137900085
// MI455X (gfx1250) — matrix operand order repaired and checked by reading, not compiled, not run on hardware
//
#include <hip/hip_runtime.h>
#include <hip/hip_bf16.h>

// ---------------------------------------------------------------------------
// Types for CDNA5 WMMA
// ---------------------------------------------------------------------------
typedef __attribute__((ext_vector_type(16))) __bf16 v16bf;
typedef __attribute__((ext_vector_type(8)))  float  v8f;

#define BT_M 128
#define BT_N 128
#define BT_K 32
#define BPAD 8           // row stride 40 bf16 = 80 B (16B-aligned rows)
#define BUFB ((BT_M) * (BT_K + BPAD) * 2)   // bytes per LDS buffer (10240)

#define FLAG_RELU   1
#define FLAG_CAUSAL 2
#define FLAG_TRANSC 4

// CDNA5 async global->LDS copy of 16 bytes (per lane), tracked by ASYNCcnt.
__device__ __forceinline__ void async_cp16(unsigned lds_off, const void* gptr)
{
    unsigned long long ga = (unsigned long long)gptr;
    asm volatile("global_load_async_to_lds_b128 %0, %1, off"
                 :: "v"(lds_off), "v"(ga) : "memory");
}
__device__ __forceinline__ void wait_async0()
{
    asm volatile("s_wait_asynccnt 0" ::: "memory");
}

// ---------------------------------------------------------------------------
// bf16 WMMA GEMM:  C = epilogue(A @ B^T_layout [+bias] [+resid])
//   A: bf16 [M,K] row-major  (per-batch stride sAb)
//   B: bf16 [N,K] row-major  ("column-major B"; per-batch stride sBb)
//   Outputs: optional f32 (Cf) and/or bf16 (Cb); FLAG_TRANSC stores Cb
//            transposed: Cb[col*ldcb + row].
//   flags: FLAG_RELU   -> relu epilogue
//          FLAG_CAUSAL -> C[i][j] = (j<=i) ? relu(v)/(i+1) : 0 (i = M index)
// Block: 256 threads = 8 waves (4x2); block tile 128x128; wave tile 32x64.
// Software-pipelined: double-buffered LDS, async copies overlap WMMA compute.
// All dims must be multiples of 128 (M,N) / 32 (K); all pointers 16B-aligned.
// ---------------------------------------------------------------------------
__global__ __launch_bounds__(256)
void k_gemm_bf16(const __bf16* __restrict__ A, long long sAb, int lda,
                 const __bf16* __restrict__ B, long long sBb, int ldb,
                 const float*  __restrict__ bias,
                 const float*  __restrict__ resid, long long sRb, int ldr,
                 float*  __restrict__ Cf, long long sCfb, int ldcf,
                 __bf16* __restrict__ Cb, long long sCbb, int ldcb,
                 int M, int N, int K, int flags)
{
    __shared__ __bf16 sA[2][BT_M][BT_K + BPAD];
    __shared__ __bf16 sB[2][BT_N][BT_K + BPAD];

    const int tid     = threadIdx.x;
    const int z       = blockIdx.z;
    const int rowTile = blockIdx.y * BT_M;
    const int colTile = blockIdx.x * BT_N;

    A += (long long)z * sAb;
    B += (long long)z * sBb;
    if (resid) resid += (long long)z * sRb;
    if (Cf)    Cf    += (long long)z * sCfb;
    if (Cb)    Cb    += (long long)z * sCbb;

    const int w     = tid >> 5;
    const int lane  = tid & 31;
    const int mBase = (w >> 1) * 32;    // 0,32,64,96
    const int nBase = (w & 1) * 64;     // 0,64
    const int hv    = lane >> 4;        // lane half
    const int l15   = lane & 15;

    // Causal: tiles strictly above the diagonal are all-zero -> write & exit.
    if ((flags & FLAG_CAUSAL) && (colTile > rowTile + BT_M - 1)) {
        #pragma unroll
        for (int mi = 0; mi < 2; ++mi)
        #pragma unroll
        for (int r = 0; r < 8; ++r) {
            int row = rowTile + mBase + mi * 16 + r + hv * 8;
            #pragma unroll
            for (int ni = 0; ni < 4; ++ni) {
                int col = colTile + nBase + ni * 16 + l15;
                if (Cb) Cb[(long long)row * ldcb + col] = (__bf16)0.f;
                if (Cf) Cf[(long long)row * ldcf + col] = 0.f;
            }
        }
        return;
    }

    // Per-thread async-copy assignments: 2 chunks of 8 bf16 for A and B each.
    int e0 = tid * 8;
    int e1 = (256 + tid) * 8;
    const int ar0 = e0 >> 5, ak0 = e0 & 31;
    const int ar1 = e1 >> 5, ak1 = e1 & 31;
    const unsigned ldsA0 = (unsigned)(size_t)&sA[0][ar0][ak0];
    const unsigned ldsA1 = (unsigned)(size_t)&sA[0][ar1][ak1];
    const unsigned ldsB0 = (unsigned)(size_t)&sB[0][ar0][ak0];
    const unsigned ldsB1 = (unsigned)(size_t)&sB[0][ar1][ak1];
    const __bf16* gA0 = A + (long long)(rowTile + ar0) * lda + ak0;
    const __bf16* gA1 = A + (long long)(rowTile + ar1) * lda + ak1;
    const __bf16* gB0 = B + (long long)(colTile + ar0) * ldb + ak0;
    const __bf16* gB1 = B + (long long)(colTile + ar1) * ldb + ak1;

    auto issue_tile = [&](int kk, int p) {
        unsigned d = p ? (unsigned)BUFB : 0u;
        async_cp16(ldsA0 + d, gA0 + kk);
        async_cp16(ldsA1 + d, gA1 + kk);
        async_cp16(ldsB0 + d, gB0 + kk);
        async_cp16(ldsB1 + d, gB1 + kk);
    };

    v8f acc[2][4];
    #pragma unroll
    for (int mi = 0; mi < 2; ++mi)
        #pragma unroll
        for (int ni = 0; ni < 4; ++ni)
            acc[mi][ni] = (v8f){};

    const int nTiles = K / BT_K;
    issue_tile(0, 0);                           // prologue: tile 0 -> buf 0

    for (int t = 0; t < nTiles; ++t) {
        wait_async0();                          // this wave's copies for tile t done
        __syncthreads();                        // whole tile t visible; buf t+1 free
        if (t + 1 < nTiles) {
            issue_tile((t + 1) * BT_K, (t + 1) & 1);   // overlap with compute
            if (t + 2 < nTiles) {
                __builtin_prefetch(gA0 + (t + 2) * BT_K, 0, 1);
                __builtin_prefetch(gB0 + (t + 2) * BT_K, 0, 1);
            }
        }

        const int p = t & 1;
        // Fragments per ISA 7.12.2 layouts.
        v16bf afrag[2], bfrag[4];
        #pragma unroll
        for (int mi = 0; mi < 2; ++mi) {
            int mr = mBase + mi * 16 + l15;
            #pragma unroll
            for (int j = 0; j < 8; ++j) {
                afrag[mi][j]     = sA[p][mr][hv * 8 + j];
                afrag[mi][j + 8] = sA[p][mr][16 + hv * 8 + j];
            }
        }
        #pragma unroll
        for (int ni = 0; ni < 4; ++ni) {
            int nc = nBase + ni * 16 + l15;
            #pragma unroll
            for (int j = 0; j < 16; ++j)
                bfrag[ni][j] = sB[p][nc][(j < 8) ? (hv * 8 + j) : (16 + hv * 8 + (j - 8))];
        }
        #pragma unroll
        for (int mi = 0; mi < 2; ++mi)
            #pragma unroll
            for (int ni = 0; ni < 4; ++ni)
                acc[mi][ni] = __builtin_amdgcn_wmma_f32_16x16x32_bf16(
                    false, afrag[mi], false, bfrag[ni],
                    (short)0, acc[mi][ni], false, false);
    }

    // Epilogue. C/D layout: VGPR r -> M=r (lanes 0-15) / M=8+r (lanes 16-31).
    #pragma unroll
    for (int mi = 0; mi < 2; ++mi)
    #pragma unroll
    for (int r = 0; r < 8; ++r) {
        int row = rowTile + mBase + mi * 16 + r + hv * 8;
        #pragma unroll
        for (int ni = 0; ni < 4; ++ni) {
            int col = colTile + nBase + ni * 16 + l15;
            float v = acc[mi][ni][r];
            if (bias)  v += bias[col];
            if (resid) v += resid[(long long)row * ldr + col];
            if (flags & FLAG_RELU) v = v > 0.f ? v : 0.f;
            if (flags & FLAG_CAUSAL) {
                if (col <= row) { v = v > 0.f ? v : 0.f; v *= 1.f / (float)(row + 1); }
                else            { v = 0.f; }
            }
            if (Cf) Cf[(long long)row * ldcf + col] = v;
            if (Cb) {
                if (flags & FLAG_TRANSC)
                    Cb[(long long)col * ldcb + row] = (__bf16)v;
                else
                    Cb[(long long)row * ldcb + col] = (__bf16)v;
            }
        }
    }
}

// ---------------------------------------------------------------------------
// emb_a[b][e] = sum_{d<640} action_flat[b][d] * W_emb[d][e]
// ---------------------------------------------------------------------------
__global__ __launch_bounds__(256)
void k_emb_a(const float* __restrict__ act, const float* __restrict__ Wemb,
             float* __restrict__ emb_a)
{
    int e = blockIdx.x * 256 + threadIdx.x;   // grid.x = 4
    int b = blockIdx.y;                       // grid.y = 16
    const float* a = act + b * 640;
    float acc = 0.f;
    for (int d = 0; d < 640; ++d) acc += a[d] * Wemb[d * 1024 + e];
    emb_a[b * 1024 + e] = acc;
}

// ---------------------------------------------------------------------------
// Hs[b,n,e] = (n even ? emb_a[b,e] : 0) + W_emb[651,e] + (n+1)*W_emb[652,e]
//           + b_emb[e] + wpe[n,e]        -> f32 and bf16 mirrors
// ---------------------------------------------------------------------------
__global__ __launch_bounds__(256)
void k_build_hs(const float* __restrict__ emb_a, const float* __restrict__ Wemb,
                const float* __restrict__ bemb, const float* __restrict__ wpe,
                float* __restrict__ hsf, __bf16* __restrict__ hsb)
{
    long long row = blockIdx.x;               // 0..16383
    int b = (int)(row >> 10), n = (int)(row & 1023);
    float pos = (float)(n + 1);
    bool even = (n & 1) == 0;
    #pragma unroll
    for (int i = 0; i < 4; ++i) {
        int e = threadIdx.x + 256 * i;
        float v = Wemb[651 * 1024 + e] + pos * Wemb[652 * 1024 + e]
                + bemb[e] + wpe[(long long)n * 1024 + e];
        if (even) v += emb_a[b * 1024 + e];
        hsf[row * 1024 + e] = v;
        hsb[row * 1024 + e] = (__bf16)v;
    }
}

// ---------------------------------------------------------------------------
// Transpose + convert weight: out[n*1024+k] = (bf16) in[k*1024+n]   (1024x1024)
// ---------------------------------------------------------------------------
__global__ __launch_bounds__(256)
void k_cvtT_bf16(const float* __restrict__ in, __bf16* __restrict__ out)
{
    int idx = blockIdx.x * 256 + threadIdx.x;     // grid = 4096
    int k = idx >> 10, n = idx & 1023;
    out[(long long)n * 1024 + k] = (__bf16)in[(long long)k * 1024 + n];
}

// ---------------------------------------------------------------------------
// Row LayerNorm over E=1024, writes f32 + bf16 mirrors
// ---------------------------------------------------------------------------
__global__ __launch_bounds__(256)
void k_layernorm(const float* __restrict__ in, const float* __restrict__ g,
                 const float* __restrict__ bta,
                 float* __restrict__ outf, __bf16* __restrict__ outb)
{
    __shared__ float s1[256], s2[256];
    long long row = blockIdx.x;
    const float* x = in + row * 1024;
    int t = threadIdx.x;
    float v[4]; float sum = 0.f, sq = 0.f;
    #pragma unroll
    for (int i = 0; i < 4; ++i) {
        v[i] = x[t + 256 * i]; sum += v[i]; sq += v[i] * v[i];
    }
    s1[t] = sum; s2[t] = sq; __syncthreads();
    for (int off = 128; off > 0; off >>= 1) {
        if (t < off) { s1[t] += s1[t + off]; s2[t] += s2[t + off]; }
        __syncthreads();
    }
    float mu  = s1[0] * (1.f / 1024.f);
    float var = s2[0] * (1.f / 1024.f) - mu * mu;
    float rs  = rsqrtf(var + 1e-5f);
    #pragma unroll
    for (int i = 0; i < 4; ++i) {
        int e = t + 256 * i;
        float y = (v[i] - mu) * rs * g[e] + bta[e];
        outf[row * 1024 + e] = y;
        outb[row * 1024 + e] = (__bf16)y;
    }
}

// ---------------------------------------------------------------------------
// preds[b,h,d] = Hs[b, 2h, :] . W_pred[:, d] + b_pred[d]   (K=1024, N=10)
// ---------------------------------------------------------------------------
__global__ __launch_bounds__(256)
void k_pred(const float* __restrict__ hs, const float* __restrict__ Wp,
            const float* __restrict__ bp, float* __restrict__ out)
{
    int idx = blockIdx.x * 256 + threadIdx.x;
    if (idx >= 16 * 512 * 10) return;
    int d = idx % 10;
    int h = (idx / 10) % 512;
    int b = idx / (10 * 512);
    long long row = (long long)b * 1024 + 2 * h;
    const float* x = hs + row * 1024;
    float acc = bp[d];
    for (int e = 0; e < 1024; ++e) acc += x[e] * Wp[e * 10 + d];
    out[idx] = acc;
}

// ---------------------------------------------------------------------------
// Host orchestration
// ---------------------------------------------------------------------------
static void launch_gemm(hipStream_t stream, dim3 grid,
                        const __bf16* A, long long sAb, int lda,
                        const __bf16* B, long long sBb, int ldb,
                        const float* bias,
                        const float* resid, long long sRb, int ldr,
                        float* Cf, long long sCfb, int ldcf,
                        __bf16* Cb, long long sCbb, int ldcb,
                        int M, int N, int K, int flags)
{
    k_gemm_bf16<<<grid, 256, 0, stream>>>(A, sAb, lda, B, sBb, ldb,
                                          bias, resid, sRb, ldr,
                                          Cf, sCfb, ldcf, Cb, sCbb, ldcb,
                                          M, N, K, flags);
}

extern "C" void kernel_launch(void* const* d_in, const int* in_sizes, int n_in,
                              void* d_out, int out_size, void* d_ws, size_t ws_size,
                              hipStream_t stream)
{
    (void)in_sizes; (void)n_in; (void)out_size; (void)ws_size;

    const float* action_set = (const float*)d_in[0];
    // d_in[1] = context_rewards: only its shape matters in the reference
    const float* wpe    = (const float*)d_in[2];
    const float* W_emb  = (const float*)d_in[3];
    const float* b_emb  = (const float*)d_in[4];
    const float* Wq     = (const float*)d_in[5];
    const float* Wk     = (const float*)d_in[6];
    const float* Wv     = (const float*)d_in[7];
    const float* ln1_g  = (const float*)d_in[8];
    const float* ln1_b  = (const float*)d_in[9];
    const float* W1     = (const float*)d_in[10];
    const float* b1     = (const float*)d_in[11];
    const float* W2     = (const float*)d_in[12];
    const float* b2     = (const float*)d_in[13];
    const float* ln2_g  = (const float*)d_in[14];
    const float* ln2_b  = (const float*)d_in[15];
    const float* W_pred = (const float*)d_in[16];
    const float* b_pred = (const float*)d_in[17];
    float* out = (float*)d_out;

    const long long BN = 16LL * 1024;      // 16384 rows total
    const long long E  = 1024;
    const long long NS = 1024;             // sequence length per batch
    const long long WELEM = E * E;

    // Workspace layout
    char* base = (char*)d_ws;
    size_t off = 0;
    auto alloc = [&](size_t bytes) -> char* {
        char* p = base + off;
        off = (off + bytes + 255) & ~(size_t)255;
        return p;
    };
    float*  hs_f32  = (float*) alloc(BN * E * 4);
    float*  tmp_f32 = (float*) alloc(BN * E * 4);
    __bf16* hs_bf   = (__bf16*)alloc(BN * E * 2);
    __bf16* q_bf    = (__bf16*)alloc(BN * E * 2);
    __bf16* k_bf    = (__bf16*)alloc(BN * E * 2);
    __bf16* vT_bf   = (__bf16*)alloc(E * BN * 2);      // [e][b*1024+j]
    __bf16* s_bf    = (__bf16*)alloc(16LL * NS * NS * 2);
    __bf16* h1_bf   = (__bf16*)alloc(BN * E * 2);
    float*  emb_a   = (float*) alloc(16LL * E * 4);
    __bf16* wqT_bf  = (__bf16*)alloc(WELEM * 2);
    __bf16* wkT_bf  = (__bf16*)alloc(WELEM * 2);
    __bf16* wvT_bf  = (__bf16*)alloc(WELEM * 2);
    __bf16* w1T_bf  = (__bf16*)alloc(WELEM * 2);
    __bf16* w2T_bf  = (__bf16*)alloc(WELEM * 2);

    // Embedding (structured-sequence shortcut) + positional build
    k_emb_a<<<dim3(4, 16), 256, 0, stream>>>(action_set, W_emb, emb_a);
    k_build_hs<<<(int)BN, 256, 0, stream>>>(emb_a, W_emb, b_emb, wpe, hs_f32, hs_bf);

    const dim3 gridBig(8, 128, 1);     // M=16384, N=1024, tiles 128x128
    const dim3 gridBat(8, 8, 16);      // per-batch M=N=1024, z=batch
    const long long SB = NS * E;       // 1024*1024 elements per batch

    for (int l = 0; l < 12; ++l) {
        // Transpose-convert this layer's weights to bf16 [N,K]
        k_cvtT_bf16<<<4096, 256, 0, stream>>>(Wq + (long long)l * WELEM, wqT_bf);
        k_cvtT_bf16<<<4096, 256, 0, stream>>>(Wk + (long long)l * WELEM, wkT_bf);
        k_cvtT_bf16<<<4096, 256, 0, stream>>>(Wv + (long long)l * WELEM, wvT_bf);
        k_cvtT_bf16<<<4096, 256, 0, stream>>>(W1 + (long long)l * WELEM, w1T_bf);
        k_cvtT_bf16<<<4096, 256, 0, stream>>>(W2 + (long long)l * WELEM, w2T_bf);

        // Q, K = Hs @ W{q,k}[l]           (bf16 outputs, row-major)
        launch_gemm(stream, gridBig, hs_bf, 0, 1024, wqT_bf, 0, 1024,
                    nullptr, nullptr, 0, 0, nullptr, 0, 0,
                    q_bf, 0, 1024, (int)BN, 1024, 1024, 0);
        launch_gemm(stream, gridBig, hs_bf, 0, 1024, wkT_bf, 0, 1024,
                    nullptr, nullptr, 0, 0, nullptr, 0, 0,
                    k_bf, 0, 1024, (int)BN, 1024, 1024, 0);
        // V = Hs @ Wv[l], stored transposed: vT[e][global_row]
        launch_gemm(stream, gridBig, hs_bf, 0, 1024, wvT_bf, 0, 1024,
                    nullptr, nullptr, 0, 0, nullptr, 0, 0,
                    vT_bf, 0, (int)BN, (int)BN, 1024, 1024, FLAG_TRANSC);

        // S = relu(Q @ K^T) * causal/(i+1)  (batched; K rows are [seq,E] = [N,K])
        launch_gemm(stream, gridBat, q_bf, SB, 1024, k_bf, SB, 1024,
                    nullptr, nullptr, 0, 0, nullptr, 0, 0,
                    s_bf, SB, 1024, 1024, 1024, 1024, FLAG_CAUSAL);

        // tmp = Hs + S @ V   (B = vT rows [e][j], per-batch offset b*1024, ldb=BN)
        launch_gemm(stream, gridBat, s_bf, SB, 1024, vT_bf, NS, (int)BN,
                    nullptr, hs_f32, SB, 1024, tmp_f32, SB, 1024,
                    nullptr, 0, 0, 1024, 1024, 1024, 0);

        // Hs = LN1(tmp)
        k_layernorm<<<(int)BN, 256, 0, stream>>>(tmp_f32, ln1_g + l * 1024,
                                                 ln1_b + l * 1024, hs_f32, hs_bf);

        // H1 = relu(Hs @ W1 + b1)
        launch_gemm(stream, gridBig, hs_bf, 0, 1024, w1T_bf, 0, 1024,
                    b1 + l * 1024, nullptr, 0, 0, nullptr, 0, 0,
                    h1_bf, 0, 1024, (int)BN, 1024, 1024, FLAG_RELU);

        // tmp = Hs + H1 @ W2 + b2
        launch_gemm(stream, gridBig, h1_bf, 0, 1024, w2T_bf, 0, 1024,
                    b2 + l * 1024, hs_f32, 0, 1024, tmp_f32, 0, 1024,
                    nullptr, 0, 0, (int)BN, 1024, 1024, 0);

        // Hs = LN2(tmp)
        k_layernorm<<<(int)BN, 256, 0, stream>>>(tmp_f32, ln2_g + l * 1024,
                                                 ln2_b + l * 1024, hs_f32, hs_bf);
    }

    // preds = Hs @ W_pred + b_pred, even rows only
    k_pred<<<(16 * 512 * 10 + 255) / 256, 256, 0, stream>>>(hs_f32, W_pred, b_pred, out);
}
